// SoftMultiHeadAttention_56968446214447
// MI455X (gfx1250) — matrix operand order repaired and checked by reading, not compiled, not run on hardware
//
#include <hip/hip_runtime.h>
#include <hip/hip_bf16.h>

// ---------------------------------------------------------------------------
// SoftMultiHeadAttention for MI455X (gfx1250, wave32, WMMA bf16 f32-accum)
// N=8, T=1024, D=512, H=8, DH=64
// ---------------------------------------------------------------------------

typedef __attribute__((ext_vector_type(16))) __bf16 v16bf;
typedef __attribute__((ext_vector_type(8)))  float  v8f;

#define TT   1024
#define NB   8
#define DD   512
#define HH   8
#define DHD  64

__device__ __forceinline__ v8f wmma_bf16(v16bf a, v16bf b, v8f c) {
  // D = A(16x32 bf16) * B(32x16 bf16) + C(16x16 f32)
  return __builtin_amdgcn_wmma_f32_16x16x32_bf16(
      /*neg_a=*/false, a, /*neg_b=*/false, b,
      /*c_mod=*/(short)0, c, /*reuse_a=*/false, /*reuse_b=*/false);
}

// Async global->LDS 16-byte copy (per-lane), tracked by ASYNCcnt.
// dst = wave-relative LDS byte offset, src = 64-bit global address.
__device__ __forceinline__ void async_copy_b128(unsigned lds_off, const void* gaddr) {
  asm volatile("global_load_async_to_lds_b128 %0, %1, off"
               :: "v"(lds_off), "v"(gaddr) : "memory");
}
__device__ __forceinline__ void wait_asynccnt0() {
  asm volatile("s_wait_asynccnt 0x0" ::: "memory");
}

// ---------------------------------------------------------------------------
// Kernel 0: W[k][n] f32 -> Wt[n][k] bf16  (512x512)
// ---------------------------------------------------------------------------
__global__ __launch_bounds__(256) void wt_transpose_kernel(
    const float* __restrict__ W, __bf16* __restrict__ Wt) {
  int idx = blockIdx.x * blockDim.x + threadIdx.x;   // 512*512 threads total
  int k = idx >> 9, n = idx & 511;
  Wt[(size_t)n * 512 + k] = (__bf16)W[(size_t)k * 512 + n];
}

// ---------------------------------------------------------------------------
// Kernel 1: C = X(f32)[8192x512] @ Wt(bf16, pre-transposed)[512x512] + bias
// Output bf16 in head-split layout:
//   mode 0: Out[h][n][t][d]   (Q and K)
//   mode 1: Out[h][n][d][t]   (V transposed for P@V B-fragments)
// Block tile 64x64, 4 waves in 2x2, each wave owns 2x2 WMMA 16x16 tiles.
// ---------------------------------------------------------------------------
__global__ __launch_bounds__(128) void qkv_gemm_kernel(
    const float* __restrict__ X, const __bf16* __restrict__ Wt,
    const float* __restrict__ bias, __bf16* __restrict__ Out, int mode) {
  __shared__ float Xs[64][32];                    // 8 KB stage (f32 -> bf16 frags)
  const int tid  = threadIdx.x;
  const int wave = tid >> 5, lane = tid & 31;
  const int wm = wave >> 1, wn = wave & 1;
  const int lrow = lane & 15, lhi = lane >> 4;
  const int tileN0 = blockIdx.x * 64;
  const int tileM0 = blockIdx.y * 64;

  v8f acc[2][2] = {};
  for (int k0 = 0; k0 < 512; k0 += 32) {
    __syncthreads();
    for (int i = tid; i < 64 * 32; i += 128) {
      int r = i >> 5, c = i & 31;
      Xs[r][c] = X[(size_t)(tileM0 + r) * 512 + k0 + c];
    }
    __syncthreads();

    v16bf afrag[2];
#pragma unroll
    for (int mi = 0; mi < 2; ++mi) {
      const float* xr = &Xs[wm * 32 + mi * 16 + lrow][0];
      const int base = lhi * 16;
#pragma unroll
      for (int j = 0; j < 8; ++j) {
        afrag[mi][j]     = (__bf16)xr[base + j];
        afrag[mi][8 + j] = (__bf16)xr[8 + base + j];
      }
    }
    v16bf bfrag[2];
#pragma unroll
    for (int ni = 0; ni < 2; ++ni) {
      int ncol = tileN0 + wn * 32 + ni * 16 + lrow;
      bfrag[ni] = *(const v16bf*)&Wt[(size_t)ncol * 512 + k0 + lhi * 16];
    }
#pragma unroll
    for (int mi = 0; mi < 2; ++mi)
#pragma unroll
      for (int ni = 0; ni < 2; ++ni)
        acc[mi][ni] = wmma_bf16(afrag[mi], bfrag[ni], acc[mi][ni]);
  }

#pragma unroll
  for (int mi = 0; mi < 2; ++mi)
#pragma unroll
    for (int ni = 0; ni < 2; ++ni) {
      const int ncol = tileN0 + wn * 32 + ni * 16 + lrow;
      const float bv = bias[ncol];
      const int h = ncol >> 6, d = ncol & 63;
#pragma unroll
      for (int j = 0; j < 8; ++j) {
        int m = tileM0 + wm * 32 + mi * 16 + j + 8 * lhi;   // C/D: M = j + 8*hi
        int n = m >> 10, t = m & 1023;
        float v = acc[mi][ni][j] + bv;
        size_t addr = (mode == 0)
            ? ((((size_t)h * NB + n) * TT + t) * DHD + d)
            : ((((size_t)h * NB + n) * DHD + d) * TT + t);
        Out[addr] = (__bf16)v;
      }
    }
}

// ---------------------------------------------------------------------------
// Kernel 2: attention. One block = (batch n, 16-query panel), 128 threads.
// LDS: S[16][1024] f32, MP[16][1024] f32 (mean-probs accum), Qs[16][64] bf16.
// Loops over all 8 heads so mean-probs needs no atomics.
// ---------------------------------------------------------------------------
__global__ __launch_bounds__(128) void attn_kernel(
    const __bf16* __restrict__ Q,     // [H][N][T][DH]
    const __bf16* __restrict__ K,     // [H][N][T][DH]
    const __bf16* __restrict__ Vt,    // [H][N][DH][T]
    const float* __restrict__ segms,  // [N][T][T]
    const int* __restrict__ mask,     // [N][T][T]
    __bf16* __restrict__ A2,          // [N*T][D] merged heads
    float* __restrict__ MPout) {      // [N][T][T]
  extern __shared__ char smem[];
  float*  S        = (float*)smem;                              // 64 KB
  float*  MP       = (float*)(smem + 16 * 1024 * 4);            // 64 KB
  __bf16* Qs       = (__bf16*)(smem + 2 * 16 * 1024 * 4);       // 2 KB
  float*  rowscale = (float*)(smem + 2 * 16 * 1024 * 4 + 2048); // 64 B

  const int tid  = threadIdx.x;
  const int wave = tid >> 5, lane = tid & 31;
  const int lrow = lane & 15, lhi = lane >> 4;
  const int q0 = blockIdx.x * 16;
  const int n  = blockIdx.y;
  const size_t rowbase = ((size_t)n * TT + q0) * TT;  // into mask/segms/MPout

  // Warm L2 with this block's mask/segms panels (64 KB each, reused 8x).
  for (int l = tid; l < 512; l += 128) {
    __builtin_prefetch(((const char*)(mask  + rowbase)) + (size_t)l * 128, 0, 1);
    __builtin_prefetch(((const char*)(segms + rowbase)) + (size_t)l * 128, 0, 1);
  }

  for (int i = tid; i < 16 * 1024; i += 128) MP[i] = 0.f;

  for (int h = 0; h < HH; ++h) {
    const __bf16* Qh = Q  + (((size_t)h * NB + n) * TT + q0) * DHD;
    const __bf16* Kh = K  +  ((size_t)h * NB + n) * TT * DHD;
    const __bf16* Vh = Vt +  ((size_t)h * NB + n) * DHD * TT;

    __syncthreads();  // previous head fully done with S/Qs
    // Async-stage the 2 KB Q panel: 128 lanes x 16 B, global -> LDS direct.
    {
      unsigned dst = (unsigned)(size_t)(smem + 2 * 16 * 1024 * 4) +
                     (unsigned)tid * 16u;
      const char* src = (const char*)Qh + (size_t)tid * 16;
      async_copy_b128(dst, src);
      wait_asynccnt0();
    }
    __syncthreads();

    // ---- S = Q @ K^T : 64 column tiles of 16, 16 per wave, K-dim 64 = 2 steps
    for (int ct = wave; ct < 64; ct += 4) {
      v8f acc = {};
#pragma unroll
      for (int ks = 0; ks < DHD; ks += 32) {
        v16bf a;
        const __bf16* qr = &Qs[lrow * DHD + ks + lhi * 16];
#pragma unroll
        for (int j = 0; j < 8; ++j) { a[j] = qr[j]; a[8 + j] = qr[8 + j]; }
        v16bf b = *(const v16bf*)&Kh[(size_t)(ct * 16 + lrow) * DHD + ks + lhi * 16];
        acc = wmma_bf16(a, b, acc);
      }
#pragma unroll
      for (int j = 0; j < 8; ++j)
        S[(j + 8 * lhi) * 1024 + ct * 16 + lrow] = acc[j];
    }
    __syncthreads();

    // ---- scale + mask
    for (int i = tid; i < 16 * 1024; i += 128) {
      int r = i >> 10, t = i & 1023;
      float s = S[i] * 0.125f;                       // DH^-0.5
      if (mask[rowbase + (size_t)r * TT + t] == 0) s = -1e9f;
      S[i] = s;
    }
    __syncthreads();

    // ---- softmax: 8 lanes per row (xor masks 1,2,4 stay in the 8-lane group)
    {
      int r = tid >> 3, sub = tid & 7;
      float mx = -3.4e38f;
      for (int t = sub; t < 1024; t += 8) mx = fmaxf(mx, S[r * 1024 + t]);
      mx = fmaxf(mx, __shfl_xor(mx, 1));
      mx = fmaxf(mx, __shfl_xor(mx, 2));
      mx = fmaxf(mx, __shfl_xor(mx, 4));
      float sum = 0.f;
      for (int t = sub; t < 1024; t += 8) {
        float e = __expf(S[r * 1024 + t] - mx);
        S[r * 1024 + t] = e;
        sum += e;
      }
      sum += __shfl_xor(sum, 1);
      sum += __shfl_xor(sum, 2);
      sum += __shfl_xor(sum, 4);
      if (sub == 0) rowscale[r] = 1.f / sum;
    }
    __syncthreads();

    // ---- P = softmax * segms ; MP += P / H
    for (int i = tid; i < 16 * 1024; i += 128) {
      int r = i >> 10, t = i & 1023;
      float p = S[i] * rowscale[r] * segms[rowbase + (size_t)r * TT + t];
      S[i] = p;
      MP[i] += p * 0.125f;                           // 1/H
    }
    __syncthreads();

    // ---- O = P @ V : each wave owns one 16-wide d-tile, K-dim = 1024
    {
      const int d0 = wave * 16;
      v8f acc = {};
      for (int kt = 0; kt < 1024; kt += 32) {
        v16bf a;
        const float* pr = &S[lrow * 1024 + kt + lhi * 16];
#pragma unroll
        for (int j = 0; j < 8; ++j) {
          a[j]     = (__bf16)pr[j];
          a[8 + j] = (__bf16)pr[8 + j];
        }
        v16bf b = *(const v16bf*)&Vh[(size_t)(d0 + lrow) * TT + kt + lhi * 16];
        acc = wmma_bf16(a, b, acc);
      }
      const int col = h * DHD + d0 + lrow;
#pragma unroll
      for (int j = 0; j < 8; ++j) {
        size_t m = (size_t)n * TT + q0 + j + 8 * lhi;
        A2[m * DD + col] = (__bf16)acc[j];
      }
    }
  }
  __syncthreads();
  for (int i = tid; i < 16 * 1024; i += 128) {
    int r = i >> 10, t = i & 1023;
    MPout[rowbase + (size_t)r * TT + t] = MP[i];
  }
}

// ---------------------------------------------------------------------------
// Kernel 3: Out(f32) = A2(bf16)[8192x512] @ WtO(bf16)[512x512]^T-layout + bo
// ---------------------------------------------------------------------------
__global__ __launch_bounds__(128) void out_gemm_kernel(
    const __bf16* __restrict__ A, const __bf16* __restrict__ Wt,
    const float* __restrict__ bias, float* __restrict__ Out) {
  const int tid  = threadIdx.x;
  const int wave = tid >> 5, lane = tid & 31;
  const int wm = wave >> 1, wn = wave & 1;
  const int lrow = lane & 15, lhi = lane >> 4;
  const int tileN0 = blockIdx.x * 64;
  const int tileM0 = blockIdx.y * 64;

  v8f acc[2][2] = {};
  for (int k0 = 0; k0 < 512; k0 += 32) {
    v16bf afrag[2];
#pragma unroll
    for (int mi = 0; mi < 2; ++mi) {
      const __bf16* ar =
          &A[(size_t)(tileM0 + wm * 32 + mi * 16 + lrow) * 512 + k0 + lhi * 16];
#pragma unroll
      for (int j = 0; j < 8; ++j) { afrag[mi][j] = ar[j]; afrag[mi][8 + j] = ar[8 + j]; }
    }
    v16bf bfrag[2];
#pragma unroll
    for (int ni = 0; ni < 2; ++ni) {
      int ncol = tileN0 + wn * 32 + ni * 16 + lrow;
      bfrag[ni] = *(const v16bf*)&Wt[(size_t)ncol * 512 + k0 + lhi * 16];
    }
#pragma unroll
    for (int mi = 0; mi < 2; ++mi)
#pragma unroll
      for (int ni = 0; ni < 2; ++ni)
        acc[mi][ni] = wmma_bf16(afrag[mi], bfrag[ni], acc[mi][ni]);
  }
#pragma unroll
  for (int mi = 0; mi < 2; ++mi)
#pragma unroll
    for (int ni = 0; ni < 2; ++ni) {
      const int ncol = tileN0 + wn * 32 + ni * 16 + lrow;
      const float bv = bias[ncol];
#pragma unroll
      for (int j = 0; j < 8; ++j) {
        int m = tileM0 + wm * 32 + mi * 16 + j + 8 * lhi;
        Out[(size_t)m * 512 + ncol] = acc[mi][ni][j] + bv;
      }
    }
}

// ---------------------------------------------------------------------------
extern "C" void kernel_launch(void* const* d_in, const int* in_sizes, int n_in,
                              void* d_out, int out_size, void* d_ws, size_t ws_size,
                              hipStream_t stream) {
  const float* q_in   = (const float*)d_in[0];
  const float* k_in   = (const float*)d_in[1];
  const float* v_in   = (const float*)d_in[2];
  const float* segms  = (const float*)d_in[3];
  const int*   mask   = (const int*)d_in[4];
  const float* Wq = (const float*)d_in[5];  const float* bq = (const float*)d_in[6];
  const float* Wk = (const float*)d_in[7];  const float* bk = (const float*)d_in[8];
  const float* Wv = (const float*)d_in[9];  const float* bv = (const float*)d_in[10];
  const float* Wo = (const float*)d_in[11]; const float* bo = (const float*)d_in[12];

  float* out   = (float*)d_out;                       // [N][T][D]
  float* mpout = out + (size_t)NB * TT * DD;          // [N][T][T]

  // Workspace carve-up (bf16): 4 x WT (256K elems) + Q,K,Vt,A2 (4M elems each)
  __bf16* WtQ = (__bf16*)d_ws;
  __bf16* WtK = WtQ + 262144;
  __bf16* WtV = WtK + 262144;
  __bf16* WtO = WtV + 262144;
  __bf16* Qb  = WtO + 262144;
  __bf16* Kb  = Qb  + (size_t)4194304;
  __bf16* Vtb = Kb  + (size_t)4194304;
  __bf16* A2  = Vtb + (size_t)4194304;

  // 0) transpose+convert weights
  wt_transpose_kernel<<<1024, 256, 0, stream>>>(Wq, WtQ);
  wt_transpose_kernel<<<1024, 256, 0, stream>>>(Wk, WtK);
  wt_transpose_kernel<<<1024, 256, 0, stream>>>(Wv, WtV);
  wt_transpose_kernel<<<1024, 256, 0, stream>>>(Wo, WtO);

  // 1) Q/K/V projections (M=8192 in 64-row tiles, N=512 in 64-col tiles)
  dim3 ggrid(DD / 64, (NB * TT) / 64);
  qkv_gemm_kernel<<<ggrid, 128, 0, stream>>>(q_in, WtQ, bq, Qb,  0);
  qkv_gemm_kernel<<<ggrid, 128, 0, stream>>>(k_in, WtK, bk, Kb,  0);
  qkv_gemm_kernel<<<ggrid, 128, 0, stream>>>(v_in, WtV, bv, Vtb, 1);

  // 2) attention (one block per (n, 16-query panel); 130 KB dynamic LDS,
  //    within the 320 KB WGP limit)
  const size_t smem_bytes = 2 * 16 * 1024 * 4 + 2048 + 64;
  (void)hipFuncSetAttribute((const void*)attn_kernel,
                            hipFuncAttributeMaxDynamicSharedMemorySize,
                            (int)smem_bytes);
  dim3 agrid(TT / 16, NB);
  attn_kernel<<<agrid, 128, smem_bytes, stream>>>(Qb, Kb, Vtb, segms, mask, A2,
                                                  mpout);

  // 3) output projection
  out_gemm_kernel<<<ggrid, 128, 0, stream>>>(A2, WtO, bo, out);
}
